// ResilienceClassifier_79413945303736
// MI455X (gfx1250) — matrix operand order repaired and checked by reading, not compiled, not run on hardware
//
#include <hip/hip_runtime.h>
#include <hip/hip_bf16.h>
#include <math.h>

typedef __attribute__((ext_vector_type(16))) __bf16 v16bf;
typedef __attribute__((ext_vector_type(8)))  float  v8f;
typedef __attribute__((ext_vector_type(4)))  float  v4f;

constexpr int BATCH = 8;
constexpr int NNODE = 4096;
constexpr int FIN   = 100;
constexpr int DIM   = 64;

// ---------------------------------------------------------------------------
// Kernel 0: zero the pooled accumulator (B*D floats) in workspace.
// ---------------------------------------------------------------------------
__global__ void zero_pool_kernel(float* __restrict__ pool) {
  int i = threadIdx.x;
  if (i < BATCH * DIM) pool[i] = 0.f;
}

// ---------------------------------------------------------------------------
// Kernel 1: ht[b, d, n] = bf16( dot(x[b,n,:], Wg[:,d]) + bg[d] )
// Transposed (d-major) so WMMA B fragments are contiguous 32B runs along K.
// Wg[f*D+d] is block-uniform (d = blockIdx.y) -> scalar loads; stores along n
// are fully coalesced.
// ---------------------------------------------------------------------------
__global__ void layer0_kernel(const float* __restrict__ x,
                              const float* __restrict__ Wg,
                              const float* __restrict__ bg,
                              __bf16* __restrict__ ht) {
  const int n = blockIdx.x * blockDim.x + threadIdx.x;
  const int d = blockIdx.y;
  const int b = blockIdx.z;
  const float* xr = x + ((size_t)b * NNODE + n) * FIN;
  float acc = bg[d];
#pragma unroll 4
  for (int f = 0; f < FIN; ++f)
    acc = fmaf(xr[f], Wg[f * DIM + d], acc);
  ht[((size_t)b * DIM + d) * NNODE + n] = (__bf16)acc;
}

// ---------------------------------------------------------------------------
// Kernel 2: For each batch, H2 = relu(adj @ h); folded directly into
// pooled[b,d] += sum_n mask[b,n] * H2[n,d]  (no H2 materialization).
//
// Wave = 16x64 output strip, 4 WMMA accumulators over K=4096 in steps of 32.
// adj is streamed once (537 MB) with NON-TEMPORAL b128 loads so the 192 MB L2
// stays dedicated to the 4 MB transposed h.  Two-deep register ping-pong
// (tiles e/o): each tile's loads are issued a full 2 tiles before its WMMAs
// consume them, and each tile buffer is refilled right after use so the
// allocator can reuse the same registers (no v_mov copies).  Prefetch is
// unconditional (speculative prefetch is droppable per ISA) to keep the loop
// a single basic block.
//
// A fragment (16x32 bf16, contiguous K): lane row = lane&15;
//   lanes 0-15 hold K 0..15, lanes 16-31 hold K 16..31 (same order as B).
// B fragment (32x16 bf16): lane column = lane&15, K-run (lane>>4)*16..+15
//   -> one contiguous 32B load per fragment from transposed h (L2-resident).
// C/D: VGPR v -> row v (lanes 0-15) or 8+v (lanes 16-31), col = lane&15.
// ---------------------------------------------------------------------------
struct KTile {
  v16bf a, b0, b1, b2, b3;
};

__device__ __forceinline__ KTile load_tile(const float* __restrict__ arow,
                                           const __bf16* __restrict__ hb,
                                           int k0, int khalf, int kq) {
  KTile t;
  // A: stream adj non-temporally (read-once data).
  v4f q0 = __builtin_nontemporal_load((const v4f*)(arow + k0 + khalf));
  v4f q1 = __builtin_nontemporal_load((const v4f*)(arow + k0 + khalf + 4));
  v4f q2 = __builtin_nontemporal_load((const v4f*)(arow + k0 + khalf + 8));
  v4f q3 = __builtin_nontemporal_load((const v4f*)(arow + k0 + khalf + 12));
  t.a[0]  = (__bf16)q0[0]; t.a[1]  = (__bf16)q0[1]; t.a[2]  = (__bf16)q0[2]; t.a[3]  = (__bf16)q0[3];
  t.a[4]  = (__bf16)q1[0]; t.a[5]  = (__bf16)q1[1]; t.a[6]  = (__bf16)q1[2]; t.a[7]  = (__bf16)q1[3];
  t.a[8]  = (__bf16)q2[0]; t.a[9]  = (__bf16)q2[1]; t.a[10] = (__bf16)q2[2]; t.a[11] = (__bf16)q2[3];
  t.a[12] = (__bf16)q3[0]; t.a[13] = (__bf16)q3[1]; t.a[14] = (__bf16)q3[2]; t.a[15] = (__bf16)q3[3];
  // B: regular (cached) loads; h stays hot in L2.
  const __bf16* bp = hb + k0 + kq;
  t.b0 = *(const v16bf*)(bp);
  t.b1 = *(const v16bf*)(bp + (size_t)16 * NNODE);
  t.b2 = *(const v16bf*)(bp + (size_t)32 * NNODE);
  t.b3 = *(const v16bf*)(bp + (size_t)48 * NNODE);
  return t;
}

#define WMMA4(T)                                                                                   \
  do {                                                                                             \
    acc0 = __builtin_amdgcn_wmma_f32_16x16x32_bf16(false, (T).a, false, (T).b0, (short)0, acc0,    \
                                                   false, false);                                  \
    acc1 = __builtin_amdgcn_wmma_f32_16x16x32_bf16(false, (T).a, false, (T).b1, (short)0, acc1,    \
                                                   false, false);                                  \
    acc2 = __builtin_amdgcn_wmma_f32_16x16x32_bf16(false, (T).a, false, (T).b2, (short)0, acc2,    \
                                                   false, false);                                  \
    acc3 = __builtin_amdgcn_wmma_f32_16x16x32_bf16(false, (T).a, false, (T).b3, (short)0, acc3,    \
                                                   false, false);                                  \
  } while (0)

__global__ void __launch_bounds__(128)
gcn_aggregate_pool_kernel(const float* __restrict__ adj,
                          const __bf16* __restrict__ ht,
                          const float* __restrict__ mask,
                          float* __restrict__ pool) {
  const int b    = blockIdx.y;
  const int m0   = blockIdx.x * 64;
  const int wave = threadIdx.x >> 5;
  const int lane = threadIdx.x & 31;
  const int half = lane >> 4;   // 0 or 1
  const int l16  = lane & 15;

  const int row = m0 + wave * 16 + l16;
  const float*  arow  = adj + ((size_t)b * NNODE + row) * NNODE;
  const int     khalf = half * 16;   // A K sub-block base
  const int     kq    = half * 16;   // B K sub-block base
  const __bf16* hb    = ht + (size_t)b * DIM * NNODE + (size_t)l16 * NNODE;

  v8f acc0 = {}, acc1 = {}, acc2 = {}, acc3 = {};

  // 2-deep ping-pong software pipeline over 128 K-tiles of 32.
  KTile e = load_tile(arow, hb, 0, khalf, kq);   // even tiles
  KTile o = load_tile(arow, hb, 32, khalf, kq);  // odd tiles
#pragma unroll 1
  for (int k0 = 64; k0 < NNODE; k0 += 64) {
    __builtin_prefetch(arow + k0 + 448 + khalf, 0, 0);  // speculative, droppable
    WMMA4(e);
    e = load_tile(arow, hb, k0, khalf, kq);
    WMMA4(o);
    o = load_tile(arow, hb, k0 + 32, khalf, kq);
  }
  WMMA4(e);
  WMMA4(o);

  // ---- ReLU + mask + pool over the wave's 16 rows ----
  const float* mrow = mask + (size_t)b * NNODE + m0 + wave * 16 + half * 8;
  float4 mA = *(const float4*)(mrow);
  float4 mB = *(const float4*)(mrow + 4);
  float mv[8] = {mA.x, mA.y, mA.z, mA.w, mB.x, mB.y, mB.z, mB.w};

  float p0 = 0.f, p1 = 0.f, p2 = 0.f, p3 = 0.f;
#pragma unroll
  for (int v = 0; v < 8; ++v) {
    p0 += fmaxf(acc0[v], 0.f) * mv[v];
    p1 += fmaxf(acc1[v], 0.f) * mv[v];
    p2 += fmaxf(acc2[v], 0.f) * mv[v];
    p3 += fmaxf(acc3[v], 0.f) * mv[v];
  }
  // lanes 0-15 and 16-31 hold the same output column -> fold halves.
  p0 += __shfl_down(p0, 16);
  p1 += __shfl_down(p1, 16);
  p2 += __shfl_down(p2, 16);
  p3 += __shfl_down(p3, 16);

  if (lane < 16) {
    float* pb = pool + b * DIM;
    atomicAdd(pb + 0 * 16 + l16, p0);
    atomicAdd(pb + 1 * 16 + l16, p1);
    atomicAdd(pb + 2 * 16 + l16, p2);
    atomicAdd(pb + 3 * 16 + l16, p3);
  }
}

// ---------------------------------------------------------------------------
// Kernel 3: head.  One wave per batch (wave32): mask-sum, normalize pooled,
// time embed, 3-layer MLP, sigmoid.
// ---------------------------------------------------------------------------
__global__ void head_kernel(const float* __restrict__ mask,
                            const float* __restrict__ t,
                            const float* __restrict__ Wt,
                            const float* __restrict__ bt,
                            const float* __restrict__ W0, const float* __restrict__ b0,
                            const float* __restrict__ W1, const float* __restrict__ b1,
                            const float* __restrict__ W2, const float* __restrict__ b2,
                            const float* __restrict__ pool,
                            float* __restrict__ out) {
  __shared__ float z[BATCH][2 * DIM];
  __shared__ float zmid[BATCH][DIM];
  const int w    = threadIdx.x >> 5;   // batch index
  const int lane = threadIdx.x & 31;

  // masked node count
  float ms = 0.f;
  for (int n = lane; n < NNODE; n += 32) ms += mask[(size_t)w * NNODE + n];
#pragma unroll
  for (int off = 16; off > 0; off >>= 1) ms += __shfl_down(ms, off);
  ms = __shfl(ms, 0);
  const float denom = fmaxf(ms, 1.f);

  for (int i = lane; i < DIM; i += 32) {
    z[w][i]       = pool[w * DIM + i] / denom;
    z[w][DIM + i] = t[w] * Wt[i] + bt[i];
  }
  __syncthreads();

  // layer 0: (2D -> D), each lane owns outputs lane and lane+32
  float a0 = b0[lane], a1 = b0[lane + 32];
  for (int k = 0; k < 2 * DIM; ++k) {
    float zk = z[w][k];
    a0 = fmaf(zk, W0[k * DIM + lane], a0);
    a1 = fmaf(zk, W0[k * DIM + lane + 32], a1);
  }
  zmid[w][lane]      = fmaxf(a0, 0.f);
  zmid[w][lane + 32] = fmaxf(a1, 0.f);
  __syncthreads();

  // layer 1: (D -> D)
  float c0 = b1[lane], c1 = b1[lane + 32];
  for (int k = 0; k < DIM; ++k) {
    float zk = zmid[w][k];
    c0 = fmaf(zk, W1[k * DIM + lane], c0);
    c1 = fmaf(zk, W1[k * DIM + lane + 32], c1);
  }
  __syncthreads();
  z[w][lane]      = fmaxf(c0, 0.f);
  z[w][lane + 32] = fmaxf(c1, 0.f);
  __syncthreads();

  // layer 2: (D -> 1) + sigmoid
  float v = z[w][lane] * W2[lane] + z[w][lane + 32] * W2[lane + 32];
#pragma unroll
  for (int off = 16; off > 0; off >>= 1) v += __shfl_down(v, off);
  if (lane == 0) out[w] = 1.f / (1.f + expf(-(v + b2[0])));
}

// ---------------------------------------------------------------------------
extern "C" void kernel_launch(void* const* d_in, const int* in_sizes, int n_in,
                              void* d_out, int out_size, void* d_ws, size_t ws_size,
                              hipStream_t stream) {
  (void)in_sizes; (void)n_in; (void)out_size; (void)ws_size;

  const float* x    = (const float*)d_in[0];
  const float* adj  = (const float*)d_in[1];
  const float* mask = (const float*)d_in[2];
  const float* t    = (const float*)d_in[3];
  const float* Wg   = (const float*)d_in[4];
  const float* bg   = (const float*)d_in[5];
  const float* Wt   = (const float*)d_in[6];
  const float* bt   = (const float*)d_in[7];
  const float* W0   = (const float*)d_in[8];
  const float* b0   = (const float*)d_in[9];
  const float* W1   = (const float*)d_in[10];
  const float* b1   = (const float*)d_in[11];
  const float* W2   = (const float*)d_in[12];
  const float* b2   = (const float*)d_in[13];

  // workspace: [ ht : B*D*N bf16 = 4 MiB ][ pooled acc : B*D floats ]
  __bf16* ht  = (__bf16*)d_ws;
  float* pool = (float*)((char*)d_ws + (size_t)BATCH * DIM * NNODE * sizeof(__bf16));

  zero_pool_kernel<<<1, 512, 0, stream>>>(pool);
  layer0_kernel<<<dim3(NNODE / 256, DIM, BATCH), 256, 0, stream>>>(x, Wg, bg, ht);
  gcn_aggregate_pool_kernel<<<dim3(NNODE / 64, BATCH), 128, 0, stream>>>(adj, ht, mask, pool);
  head_kernel<<<1, BATCH * 32, 0, stream>>>(mask, t, Wt, bt, W0, b0, W1, b1, W2, b2,
                                            pool, (float*)d_out);
}
